// EpisodicMemory_81535659147882
// MI455X (gfx1250) — matrix operand order repaired and checked by reading, not compiled, not run on hardware
//
#include <hip/hip_runtime.h>
#include <math.h>

// ---------------- CDNA5 WMMA types / helpers (wave32) ----------------
typedef __bf16 bf16_t;
typedef __attribute__((ext_vector_type(16))) __bf16 v16bf;
typedef __attribute__((ext_vector_type(8)))  float  v8f;

#define BS     8
#define NSEQ   1024
#define BH     4
#define DIM    128
#define MSLOT  2048
#define CCAND  64
#define KRET   32
#define NEGV   (-1e9f)
#define SMAXV  4.0f
#define BUDGET 256.0f

__device__ __forceinline__ v8f wmma_bf16(v16bf a, v16bf b, v8f c) {
  // D = A(16x32 bf16) * B(32x16 bf16) + C(16x16 f32)
  return __builtin_amdgcn_wmma_f32_16x16x32_bf16(false, a, false, b, (short)0, c,
                                                 false, false);
}

// A-matrix fragment (16x32 bf16) from bf16 LDS: lane&15 = row,
// lanes0-15 hold K={0..15}, lanes16-31 K={16..31}.
__device__ __forceinline__ v16bf load_a16(const bf16_t* src, int pitch, int k0, int lane) {
  const int row = lane & 15;
  const int kb  = (lane >> 4) * 16;
  const bf16_t* p = src + row * pitch + k0;
  v16bf a;
#pragma unroll
  for (int e = 0; e < 8; ++e) a[e] = p[kb + e];
#pragma unroll
  for (int e = 0; e < 8; ++e) a[8 + e] = p[8 + kb + e];
  return a;
}

// Same A fragment, sourced from f32 global memory (convert on load).
__device__ __forceinline__ v16bf load_a16f(const float* src, int pitch, int k0, int lane) {
  const int row = lane & 15;
  const int kb  = (lane >> 4) * 16;
  const float* p = src + (long)row * pitch + k0;
  v16bf a;
#pragma unroll
  for (int e = 0; e < 8; ++e) a[e] = (bf16_t)p[kb + e];
#pragma unroll
  for (int e = 0; e < 8; ++e) a[8 + e] = (bf16_t)p[8 + kb + e];
  return a;
}

// B-matrix fragment (32x16): element (k,n) from src[n*pitch + k]  ([N][K] source,
// contiguous in k -> lowers to 2x ds_load_b128 per lane)
__device__ __forceinline__ v16bf load_b_nk(const bf16_t* src, int pitch, int k0, int lane) {
  const int n  = lane & 15;
  const int kb = k0 + (lane >> 4) * 16;
  const bf16_t* p = src + n * pitch + kb;
  v16bf b;
#pragma unroll
  for (int e = 0; e < 16; ++e) b[e] = p[e];
  return b;
}

// C/D fragment (16x16 f32): lane&15 = col, vgpr r -> row r + 8*(lane>=16)
__device__ __forceinline__ void store_c(float* dst, int pitch, v8f c, int lane) {
  const int n  = lane & 15;
  const int rb = (lane >> 4) * 8;
#pragma unroll
  for (int r = 0; r < 8; ++r) dst[(rb + r) * pitch + n] = c[r];
}

// ---- CDNA5 async global->LDS copy (ASYNCcnt-tracked), inline asm ----
__device__ __forceinline__ void async_copy_b128(unsigned lds_addr, const void* gaddr) {
  asm volatile("global_load_async_to_lds_b128 %0, %1, off"
               :: "v"(lds_addr), "v"((unsigned long long)(uintptr_t)gaddr)
               : "memory");
}
__device__ __forceinline__ void wait_async0() {
  asm volatile("s_wait_asynccnt 0x0" ::: "memory");
}

// =====================================================================
// Kernel A: read attention (top-32 thresholded softmax) + novelty
// One block = 64 query rows of one (x,b). Two passes over M (recompute).
// q/q_nov live as per-wave register fragments; em_K/em_V tiles arrive
// via GLOBAL_LOAD_ASYNC_TO_LDS_B128 into f32 staging, then bf16 convert.
// =====================================================================
#define TSN 64
#define TSM 32

__global__ __launch_bounds__(256) void em_read_novelty_kernel(
    const float* __restrict__ q, const float* __restrict__ q_nov,
    const float* __restrict__ surprise, const float* __restrict__ w_nov,
    const float* __restrict__ em_K, const float* __restrict__ em_V,
    const float* __restrict__ em_S,
    float* __restrict__ read_out, float* __restrict__ novelty) {
  const int nt = blockIdx.x, b = blockIdx.y, x = blockIdx.z;
  const int tid = threadIdx.x, lane = tid & 31, wv = tid >> 5;
  const int n0 = nt * TSN;
  const int pair = x * BH + b;

  __shared__ float  s_stA[TSM][DIM];  // 16 KB  async staging (em_K)
  __shared__ float  s_stB[TSM][DIM];  // 16 KB  async staging (em_V)
  __shared__ bf16_t s_K [TSM][DIM];   //  8 KB
  __shared__ float  s_sc [TSN][TSM];  //  8 KB
  __shared__ float  s_sim[TSN][TSM];  //  8 KB (pass 2: bf16 V^T [DIM][TSM])
  __shared__ bf16_t s_p [TSN][TSM];   //  4 KB
  __shared__ float  s_S[TSM];
  __shared__ float  s_thr[TSN], s_rmax[TSN], s_rsum[TSN];

  const unsigned stA = (unsigned)(uintptr_t)&s_stA[0][0];
  const unsigned stB = (unsigned)(uintptr_t)&s_stB[0][0];

  // per-wave A fragments for this wave's 16 query rows (rows 16*(wv>>1))
  const int ri = wv >> 1, cj = wv & 1, djb = (wv & 1) * 4;
  v16bf qf[4], qnf[4];
  {
    const float* qrow  = q     + ((long)(x * NSEQ + n0 + 16 * ri) * BH + b) * DIM;
    const float* qnrow = q_nov + ((long)(x * NSEQ + n0 + 16 * ri) * BH + b) * DIM;
#pragma unroll
    for (int kk = 0; kk < 4; ++kk) {
      qf[kk]  = load_a16f(qrow,  BH * DIM, kk * 32, lane);
      qnf[kk] = load_a16f(qnrow, BH * DIM, kk * 32, lane);
    }
  }

  // ---------- pass 1: scores & sim; streaming top-K and max-sim ----------
  float topk[KRET];
#pragma unroll
  for (int j = 0; j < KRET; ++j) topk[j] = NEGV;
  float tmin = NEGV; int tpos = 0;
  float rmax = NEGV, msim = -1.0f;

  for (int mc = 0; mc < MSLOT; mc += TSM) {
    {  // async DMA: em_K chunk (16 KB = 1024 x b128, 4 per thread)
      const float* gK = em_K + ((long)pair * MSLOT + mc) * DIM;
#pragma unroll
      for (int j = 0; j < 4; ++j) {
        const int seg = tid + j * 256;
        async_copy_b128(stA + seg * 16, gK + seg * 4);
      }
    }
    if (tid < TSM) s_S[tid] = em_S[(long)pair * MSLOT + mc + tid];
    wait_async0();
    __syncthreads();
    for (int i = tid; i < TSM * DIM; i += 256)
      s_K[i >> 7][i & (DIM - 1)] = (bf16_t)s_stA[i >> 7][i & (DIM - 1)];
    __syncthreads();

    {  // 64x32 score + sim tiles: 8 subtiles, one per wave, K=128
      v8f accS = (v8f){0,0,0,0,0,0,0,0};
      v8f accN = (v8f){0,0,0,0,0,0,0,0};
#pragma unroll
      for (int kk = 0; kk < 4; ++kk) {
        const v16bf bfr = load_b_nk(&s_K[16 * cj][0], DIM, kk * 32, lane);
        accS = wmma_bf16(qf[kk],  bfr, accS);
        accN = wmma_bf16(qnf[kk], bfr, accN);
      }
      store_c(&s_sc [16 * ri][16 * cj], TSM, accS, lane);
      store_c(&s_sim[16 * ri][16 * cj], TSM, accN, lane);
    }
    __syncthreads();

    if (tid < TSN) {
      for (int ml = 0; ml < TSM; ++ml) {
        const bool act = s_S[ml] > 0.0f;
        const float sv = act ? s_sc[tid][ml] : NEGV;
        rmax = fmaxf(rmax, sv);
        if (sv > tmin) {                // replace current min of top-32
          topk[tpos] = sv;
          tmin = topk[0]; tpos = 0;
#pragma unroll
          for (int j = 1; j < KRET; ++j)
            if (topk[j] < tmin) { tmin = topk[j]; tpos = j; }
        }
        const float sm = act ? s_sim[tid][ml] : -1.0f;
        msim = fmaxf(msim, sm);
      }
    }
    __syncthreads();
  }

  if (tid < TSN) {
    s_thr[tid]  = tmin;   // 32nd largest (threshold)
    s_rmax[tid] = rmax;
    const int sidx = (x * NSEQ + n0 + tid) * BH + b;
    const float wn = w_nov[sidx];
    novelty[sidx] = wn * surprise[sidx] + (1.0f - wn) * (1.0f - fmaxf(msim, 0.0f));
  }
  __syncthreads();

  // ---------- pass 2: recompute scores, threshold+exp, p @ V ----------
  bf16_t* s_Vt = (bf16_t*)&s_sim[0][0];   // V^T [DIM][TSM], overlays s_sim
  v8f o[4];
#pragma unroll
  for (int s = 0; s < 4; ++s) o[s] = (v8f){0,0,0,0,0,0,0,0};
  float rsum = 0.0f;

  for (int mc = 0; mc < MSLOT; mc += TSM) {
    {  // async DMA: em_K -> stA, em_V -> stB
      const long cb = ((long)pair * MSLOT + mc) * DIM;
      const float* gK = em_K + cb;
      const float* gV = em_V + cb;
#pragma unroll
      for (int j = 0; j < 4; ++j) {
        const int seg = tid + j * 256;
        async_copy_b128(stA + seg * 16, gK + seg * 4);
        async_copy_b128(stB + seg * 16, gV + seg * 4);
      }
    }
    if (tid < TSM) s_S[tid] = em_S[(long)pair * MSLOT + mc + tid];
    wait_async0();
    __syncthreads();
    for (int i = tid; i < TSM * DIM; i += 256) {
      const int m = i >> 7, d = i & (DIM - 1);
      s_K[m][d]           = (bf16_t)s_stA[m][d];
      s_Vt[d * TSM + m]   = (bf16_t)s_stB[m][d];   // transpose for B fragments
    }
    __syncthreads();

    {
      v8f accS = (v8f){0,0,0,0,0,0,0,0};
#pragma unroll
      for (int kk = 0; kk < 4; ++kk) {
        const v16bf bfr = load_b_nk(&s_K[16 * cj][0], DIM, kk * 32, lane);
        accS = wmma_bf16(qf[kk], bfr, accS);
      }
      store_c(&s_sc[16 * ri][16 * cj], TSM, accS, lane);
    }
    __syncthreads();

    if (tid < TSN) {
      const float th = s_thr[tid], rm = s_rmax[tid];
      for (int ml = 0; ml < TSM; ++ml) {
        const bool act = s_S[ml] > 0.0f;
        const float sv = act ? s_sc[tid][ml] : NEGV;
        const float p  = (sv >= th) ? __expf(sv - rm) : 0.0f;
        s_p[tid][ml] = (bf16_t)p;
        rsum += p;
      }
    }
    __syncthreads();

    {  // out[64x128] += p[64x32] @ V[32x128]; 4 subtiles per wave
      const v16bf a = load_a16(&s_p[16 * ri][0], TSM, 0, lane);
#pragma unroll
      for (int s = 0; s < 4; ++s) {
        const v16bf bv = load_b_nk(&s_Vt[(16 * (djb + s)) * TSM], TSM, 0, lane);
        o[s] = wmma_bf16(a, bv, o[s]);
      }
    }
    __syncthreads();
  }

  if (tid < TSN) s_rsum[tid] = rsum;
  __syncthreads();

#pragma unroll
  for (int s = 0; s < 4; ++s) {
    const int dcol = 16 * (djb + s) + (lane & 15);
#pragma unroll
    for (int r = 0; r < 8; ++r) {
      const int nloc = 16 * ri + (lane >> 4) * 8 + r;
      const float val = o[s][r] / s_rsum[nloc];
      read_out[((long)(x * NSEQ + n0 + nloc) * BH + b) * DIM + dcol] = val;
    }
  }
}

// =====================================================================
// Kernel B: top-64 novelty candidates per (x,b) (iterative argmax)
// =====================================================================
__global__ __launch_bounds__(256) void em_topc_kernel(
    const float* __restrict__ novelty, int* __restrict__ cidx,
    float* __restrict__ cscore) {
  const int b = blockIdx.x, x = blockIdx.y, pair = x * BH + b;
  const int tid = threadIdx.x;
  __shared__ float sv[NSEQ];
  __shared__ float rv[256];
  __shared__ int   rix[256];
  for (int i = tid; i < NSEQ; i += 256) sv[i] = novelty[(x * NSEQ + i) * BH + b];
  __syncthreads();
  for (int c = 0; c < CCAND; ++c) {
    float bv = -INFINITY; int bi = NSEQ;
    for (int i = tid; i < NSEQ; i += 256) {
      const float v = sv[i];
      if (v > bv || (v == bv && i < bi)) { bv = v; bi = i; }
    }
    rv[tid] = bv; rix[tid] = bi;
    __syncthreads();
    for (int off = 128; off > 0; off >>= 1) {
      if (tid < off) {
        const float v2 = rv[tid + off]; const int i2 = rix[tid + off];
        if (v2 > rv[tid] || (v2 == rv[tid] && i2 < rix[tid])) { rv[tid] = v2; rix[tid] = i2; }
      }
      __syncthreads();
    }
    if (tid == 0) {
      cidx[pair * CCAND + c]  = rix[0];
      cscore[pair * CCAND + c] = rv[0];
      sv[rix[0]] = -INFINITY;
    }
    __syncthreads();
  }
}

// =====================================================================
// Kernel C1: gather/normalize candidates, slot-score GEMM, temperature
// softmax over M, alpha & a_slot, budget scale.  One block per (x,b).
// =====================================================================
__global__ __launch_bounds__(256) void em_write_alpha_kernel(
    const float* __restrict__ q_nov, const float* __restrict__ v_nov,
    const float* __restrict__ g_em, const float* __restrict__ tau,
    const float* __restrict__ ww, const float* __restrict__ em_K,
    const float* __restrict__ em_S,
    const int* __restrict__ cidx, const float* __restrict__ cscore,
    float* __restrict__ candK, float* __restrict__ candV,
    float* __restrict__ alpha, float* __restrict__ aslot,
    float* __restrict__ scalep) {
  const int b = blockIdx.x, x = blockIdx.y, pair = x * BH + b;
  const int tid = threadIdx.x, lane = tid & 31, wv = tid >> 5;

  __shared__ bf16_t s_cK[CCAND][DIM];  // 16 KB
  __shared__ bf16_t s_K[64][DIM];      // 16 KB
  __shared__ float  s_sc[CCAND][64];   // 16 KB
  __shared__ float  s_cw[CCAND], s_nrm[CCAND], s_rmax[CCAND], s_rsum[CCAND];
  __shared__ float  s_red[CCAND][4];
  __shared__ int    s_ix[CCAND];
  __shared__ float  s_tmp[CCAND];
  __shared__ float  s_sum[1];

  if (tid < CCAND) { s_ix[tid] = cidx[pair * CCAND + tid]; s_tmp[tid] = cscore[pair * CCAND + tid]; }
  __syncthreads();
  if (tid == 0) {
    float s = 0.0f;
    for (int c = 0; c < CCAND; ++c) s += s_tmp[c];
    s_sum[0] = s + 1e-8f;
  }
  __syncthreads();
  if (tid < CCAND) s_cw[tid] = s_tmp[tid] / s_sum[0];

  const int c = tid & (CCAND - 1), part = tid >> 6;  // 4 threads per candidate
  {  // gather cand_K (normalized) and cand_V
    const long base = ((long)(x * NSEQ + s_ix[c]) * BH + b) * DIM;
    float ss = 0.0f;
    for (int d = part * 32; d < part * 32 + 32; ++d) { const float v = q_nov[base + d]; ss += v * v; }
    s_red[c][part] = ss;
    __syncthreads();
    if (part == 0)
      s_nrm[c] = fmaxf(sqrtf(s_red[c][0] + s_red[c][1] + s_red[c][2] + s_red[c][3]), 1e-12f);
    __syncthreads();
    for (int d = part * 32; d < part * 32 + 32; ++d) {
      const float kn = q_nov[base + d] / s_nrm[c];
      candK[((long)pair * CCAND + c) * DIM + d] = kn;
      s_cK[c][d] = (bf16_t)kn;
      candV[((long)pair * CCAND + c) * DIM + d] = v_nov[base + d];
    }
  }
  __syncthreads();

  // slot scores [64 x 2048] via WMMA; pass 1: raw scores + row max
  const float www = ww[pair];
  float rmax = -INFINITY;
  for (int mc = 0; mc < MSLOT; mc += 64) {
    for (int i = tid; i < 64 * DIM; i += 256) {
      const int m = i >> 7, d = i & (DIM - 1);
      s_K[m][d] = (bf16_t)em_K[((long)pair * MSLOT + mc + m) * DIM + d];
    }
    if (mc + 64 < MSLOT)
      __builtin_prefetch(&em_K[((long)pair * MSLOT + mc + 64) * DIM + (tid & 127)], 0, 1);
    __syncthreads();
#pragma unroll
    for (int t = 0; t < 2; ++t) {  // 16 subtiles / 8 waves
      const int sub = wv * 2 + t, ri = sub >> 2, cj = sub & 3;
      v8f acc = (v8f){0,0,0,0,0,0,0,0};
#pragma unroll
      for (int kk = 0; kk < 4; ++kk) {
        const v16bf a  = load_a16(&s_cK[16 * ri][0], DIM, kk * 32, lane);
        const v16bf bb = load_b_nk(&s_K[16 * cj][0], DIM, kk * 32, lane);
        acc = wmma_bf16(a, bb, acc);
      }
      store_c(&s_sc[16 * ri][16 * cj], 64, acc, lane);
    }
    __syncthreads();
    if (tid < CCAND) {
      for (int ml = 0; ml < 64; ++ml) {
        const int m = mc + ml;
        const float v = s_sc[tid][ml] - www * em_S[(long)pair * MSLOT + m];
        alpha[((long)pair * CCAND + tid) * MSLOT + m] = v;
        rmax = fmaxf(rmax, v);
      }
    }
    __syncthreads();
  }
  if (tid < CCAND) s_rmax[tid] = rmax;
  __syncthreads();

  // pass 2: exp((s - rowmax)/tau), row sums
  const float tauc = fmaxf(tau[pair], 0.01f);
  {
    float ss = 0.0f;
    const long rb = ((long)pair * CCAND + c) * MSLOT;
    for (int m = part * (MSLOT / 4); m < (part + 1) * (MSLOT / 4); ++m) {
      const float p = __expf((alpha[rb + m] - s_rmax[c]) / tauc);
      alpha[rb + m] = p;
      ss += p;
    }
    s_red[c][part] = ss;
    __syncthreads();
    if (part == 0) s_rsum[c] = s_red[c][0] + s_red[c][1] + s_red[c][2] + s_red[c][3];
  }
  __syncthreads();

  // pass 3: alpha = g * cand_w * slot_w; a_slot = sum_c alpha; budget total
  const float g = g_em[pair];
  float tot = 0.0f;
  for (int m = tid; m < MSLOT; m += 256) {
    float acc = 0.0f;
    const long base = (long)pair * CCAND * MSLOT + m;
    for (int cc = 0; cc < CCAND; ++cc) {
      const long idx = base + (long)cc * MSLOT;
      const float al = g * s_cw[cc] * alpha[idx] / s_rsum[cc];
      alpha[idx] = al;
      acc += al;
    }
    aslot[(long)pair * MSLOT + m] = acc;
    tot += fminf(em_S[(long)pair * MSLOT + m] + acc, SMAXV);
  }
  float* red = &s_red[0][0];  // reuse as 256-wide reduction buffer
  red[tid] = tot;
  __syncthreads();
  for (int off = 128; off > 0; off >>= 1) {
    if (tid < off) red[tid] += red[tid + off];
    __syncthreads();
  }
  if (tid == 0) scalep[pair] = fminf(1.0f, BUDGET / (red[0] + 1e-8f));
}

// =====================================================================
// Kernel C2: blended_K/V = alpha^T @ cand (WMMA), EMA update, budget.
// One block = 64 slots of one (x,b). B operands staged transposed so
// fragments are contiguous (b128 LDS loads).
// =====================================================================
__global__ __launch_bounds__(256) void em_write_update_kernel(
    const float* __restrict__ em_K, const float* __restrict__ em_V,
    const float* __restrict__ em_S, const float* __restrict__ em_age,
    const float* __restrict__ candK, const float* __restrict__ candV,
    const float* __restrict__ alpha, const float* __restrict__ aslot,
    const float* __restrict__ scalep,
    float* __restrict__ outK, float* __restrict__ outV,
    float* __restrict__ outS, float* __restrict__ outAge) {
  const int mt = blockIdx.x, b = blockIdx.y, x = blockIdx.z;
  const int pair = x * BH + b, m0 = mt * 64;
  const int tid = threadIdx.x, lane = tid & 31, wv = tid >> 5;

  __shared__ __align__(16) char smem[45 * 1024];
  bf16_t* s_A   = (bf16_t*)smem;                     // [64m][64c]  8 KB (alpha^T)
  bf16_t* s_B1t = (bf16_t*)(smem + 8 * 1024);        // candK^T [DIM][CCAND] 16 KB
  bf16_t* s_B2t = (bf16_t*)(smem + 24 * 1024);       // candV^T [DIM][CCAND] 16 KB
  float*  s_bK  = (float*)(smem + 8 * 1024);         // overlay after GEMM: 32 KB
  float*  s_as  = (float*)(smem + 40 * 1024);        // a_slot [64]
  float*  s_nr  = (float*)(smem + 40 * 1024 + 256);  // norms  [64]
  float*  s_rd  = (float*)(smem + 40 * 1024 + 512);  // [64][4]

  for (int i = tid; i < 64 * CCAND; i += 256) {
    const int ml = i >> 6, cc = i & (CCAND - 1);
    s_A[ml * CCAND + cc] = (bf16_t)alpha[((long)pair * CCAND + cc) * MSLOT + m0 + ml];
  }
  for (int i = tid; i < CCAND * DIM; i += 256) {
    const int cc = i >> 7, d = i & (DIM - 1);
    s_B1t[d * CCAND + cc] = (bf16_t)candK[(long)pair * CCAND * DIM + i];
    s_B2t[d * CCAND + cc] = (bf16_t)candV[(long)pair * CCAND * DIM + i];
  }
  if (tid < 64) s_as[tid] = aslot[(long)pair * MSLOT + m0 + tid];
  const float scal = scalep[pair];
  __syncthreads();

  const int ri = wv >> 1, djb = (wv & 1) * 4;
  v8f oK[4], oV[4];
#pragma unroll
  for (int s = 0; s < 4; ++s) { oK[s] = (v8f){0,0,0,0,0,0,0,0}; oV[s] = oK[s]; }
  const v16bf a0 = load_a16(s_A + 16 * ri * CCAND, CCAND, 0, lane);
  const v16bf a1 = load_a16(s_A + 16 * ri * CCAND, CCAND, 32, lane);
#pragma unroll
  for (int s = 0; s < 4; ++s) {
    const bf16_t* b1 = s_B1t + (16 * (djb + s)) * CCAND;
    const bf16_t* b2 = s_B2t + (16 * (djb + s)) * CCAND;
    oK[s] = wmma_bf16(a0, load_b_nk(b1, CCAND, 0,  lane), oK[s]);
    oK[s] = wmma_bf16(a1, load_b_nk(b1, CCAND, 32, lane), oK[s]);
    oV[s] = wmma_bf16(a0, load_b_nk(b2, CCAND, 0,  lane), oV[s]);
    oV[s] = wmma_bf16(a1, load_b_nk(b2, CCAND, 32, lane), oV[s]);
  }

  // V: EMA update straight from fragments
#pragma unroll
  for (int s = 0; s < 4; ++s) {
    const int dcol = 16 * (djb + s) + (lane & 15);
#pragma unroll
    for (int r = 0; r < 8; ++r) {
      const int mloc = 16 * ri + (lane >> 4) * 8 + r;
      const float a = s_as[mloc];
      const long gi = ((long)pair * MSLOT + m0 + mloc) * DIM + dcol;
      const float old = em_V[gi];
      float nv = old;
      if (a > 1e-8f) {
        const float ae = fminf(a, 1.0f);
        const float bv = oV[s][r] / fmaxf(a, 1e-8f);
        nv = (1.0f - ae) * old + ae * bv;
      }
      outV[gi] = nv;
    }
  }
  __syncthreads();

  // K: stage for row norms (overlay), then EMA with unit()
#pragma unroll
  for (int s = 0; s < 4; ++s)
    store_c(s_bK + (16 * ri) * DIM + 16 * (djb + s), DIM, oK[s], lane);
  __syncthreads();
  {
    const int row = tid & 63, p4 = tid >> 6;
    float ss = 0.0f;
    for (int d = p4 * 32; d < p4 * 32 + 32; ++d) { const float v = s_bK[row * DIM + d]; ss += v * v; }
    s_rd[row * 4 + p4] = ss;
    __syncthreads();
    if (p4 == 0)
      s_nr[row] = fmaxf(
          sqrtf(s_rd[row * 4] + s_rd[row * 4 + 1] + s_rd[row * 4 + 2] + s_rd[row * 4 + 3]),
          1e-12f);
  }
  __syncthreads();
  for (int i = tid; i < 64 * DIM; i += 256) {
    const int mloc = i >> 7, d = i & (DIM - 1);
    const float a = s_as[mloc];
    const long gi = ((long)pair * MSLOT + m0 + mloc) * DIM + d;
    const float old = em_K[gi];
    float nk = old;
    if (a > 1e-8f) {
      const float ae = fminf(a, 1.0f);
      nk = (1.0f - ae) * old + ae * (s_bK[i] / s_nr[mloc]);  // unit() is scale-invariant
    }
    outK[gi] = nk;
  }
  if (tid < 64) {
    const long gm = (long)pair * MSLOT + m0 + tid;
    const float a = s_as[tid];
    outS[gm]   = fmaxf(fminf(em_S[gm] + a, SMAXV), 0.0f) * scal;
    outAge[gm] = em_age[gm] * (1.0f - a);
  }
}

// =====================================================================
extern "C" void kernel_launch(void* const* d_in, const int* in_sizes, int n_in,
                              void* d_out, int out_size, void* d_ws, size_t ws_size,
                              hipStream_t stream) {
  (void)in_sizes; (void)n_in; (void)out_size; (void)ws_size;
  const float* q        = (const float*)d_in[0];
  const float* q_nov    = (const float*)d_in[1];
  const float* v_nov    = (const float*)d_in[2];
  const float* surprise = (const float*)d_in[3];
  const float* w_nov    = (const float*)d_in[4];
  const float* g_em     = (const float*)d_in[5];
  const float* tau      = (const float*)d_in[6];
  const float* ww       = (const float*)d_in[7];
  const float* em_K     = (const float*)d_in[8];
  const float* em_V     = (const float*)d_in[9];
  const float* em_S     = (const float*)d_in[10];
  const float* em_age   = (const float*)d_in[11];
  // d_in[12] = C_cand scalar (== 64 in setup); baked in as CCAND.

  float* out      = (float*)d_out;
  float* read_out = out;                                        // [8,1024,4,128]
  float* novelty  = read_out + (size_t)BS * NSEQ * BH * DIM;    // [8,1024,4]
  float* outK     = novelty + (size_t)BS * NSEQ * BH;           // [8,4,2048,128]
  float* outV     = outK + (size_t)BS * BH * MSLOT * DIM;
  float* outS     = outV + (size_t)BS * BH * MSLOT * DIM;       // [8,4,2048]
  float* outAge   = outS + (size_t)BS * BH * MSLOT;

  // workspace layout (~19.2 MB of f32)
  int*   w_idx   = (int*)d_ws;                                  // [32][64]
  float* wf      = (float*)d_ws;
  float* w_cs    = wf + 2048;                                   // [32][64]
  float* w_candK = wf + 4096;                                   // [32][64][128]
  float* w_candV = w_candK + (size_t)32 * CCAND * DIM;
  float* w_alpha = w_candV + (size_t)32 * CCAND * DIM;          // [32][64][2048]
  float* w_aslot = w_alpha + (size_t)32 * CCAND * MSLOT;        // [32][2048]
  float* w_scale = w_aslot + (size_t)32 * MSLOT;                // [32]

  em_read_novelty_kernel<<<dim3(NSEQ / TSN, BH, BS), 256, 0, stream>>>(
      q, q_nov, surprise, w_nov, em_K, em_V, em_S, read_out, novelty);
  em_topc_kernel<<<dim3(BH, BS), 256, 0, stream>>>(novelty, w_idx, w_cs);
  em_write_alpha_kernel<<<dim3(BH, BS), 256, 0, stream>>>(
      q_nov, v_nov, g_em, tau, ww, em_K, em_S, w_idx, w_cs,
      w_candK, w_candV, w_alpha, w_aslot, w_scale);
  em_write_update_kernel<<<dim3(MSLOT / 64, BH, BS), 256, 0, stream>>>(
      em_K, em_V, em_S, em_age, w_candK, w_candV, w_alpha, w_aslot, w_scale,
      outK, outV, outS, outAge);
}
